// GateFusion_23115513987443
// MI455X (gfx1250) — matrix operand order repaired and checked by reading, not compiled, not run on hardware
//
#include <hip/hip_runtime.h>
#include <math.h>

typedef __attribute__((ext_vector_type(16))) __bf16          v16bf;
typedef __attribute__((ext_vector_type(2)))  __bf16          v2bf;
typedef __attribute__((ext_vector_type(16))) unsigned short  v16u;
typedef __attribute__((ext_vector_type(8)))  float           v8f;
typedef __attribute__((ext_vector_type(16))) float           v16f;
typedef __attribute__((ext_vector_type(2)))  float           v2f;

#define LN_EPS 1e-5f

// Build a 16x32 bf16 A-fragment directly from a global f32 row pointer.
// ISA A layout: lane l holds row M=l&15; half-wave g=l>>4 selects K runs
// [g*8, g*8+8) and [16+g*8, 16+g*8+8) within the 32-wide K chunk.
__device__ __forceinline__ v16bf load_a_chunk(const float* __restrict__ rowp,
                                              int kbase, int g) {
    const float* p0 = rowp + kbase + g * 8;
    const float* p1 = rowp + kbase + 16 + g * 8;
    float4 x0 = *(const float4*)(p0);
    float4 x1 = *(const float4*)(p0 + 4);
    float4 y0 = *(const float4*)(p1);
    float4 y1 = *(const float4*)(p1 + 4);
    v16f f;
    f[0]  = x0.x; f[1]  = x0.y; f[2]  = x0.z; f[3]  = x0.w;
    f[4]  = x1.x; f[5]  = x1.y; f[6]  = x1.z; f[7]  = x1.w;
    f[8]  = y0.x; f[9]  = y0.y; f[10] = y0.z; f[11] = y0.w;
    f[12] = y1.x; f[13] = y1.y; f[14] = y1.z; f[15] = y1.w;
    return __builtin_convertvector(f, v16bf);   // RNE; hw packed cvt if available
}

#define WMMA_BF16(A, B, C) \
    __builtin_amdgcn_wmma_f32_16x16x32_bf16(false, (A), false, (B), (short)0, (C), false, false)

// LDS tile map (each tile: 32K x 16N bf16, 512 elems, lane-contiguous):
//   tiles  0..19 : W_gate1  (kt 0..4, nt 0..3)  -> tile = kt*4 + nt
//   tiles 20..35 : W_detail (kt 0..1, nt 0..7)  -> tile = 20 + kt*8 + nt
//   tiles 36..59 : W_context(kt 0..2, nt 0..7)  -> tile = 36 + kt*8 + nt
#define N_TILES 60
#define ROW_TILES_PER_WAVE 4            // 8 waves * 4 tiles * 16 rows = 512 rows/block
#define ROWS_PER_BLOCK 512

__global__ __launch_bounds__(256, 1)
void gate_fusion_wmma_kernel(const float* __restrict__ xd,
                             const float* __restrict__ xc,
                             const float* __restrict__ Wd,
                             const float* __restrict__ Wc,
                             const float* __restrict__ Wg1,
                             const float* __restrict__ gamma,
                             const float* __restrict__ beta,
                             const float* __restrict__ Wg2,
                             float* __restrict__ out,
                             int Ntot)
{
    __shared__ unsigned short wsm[N_TILES * 512] __attribute__((aligned(32)));

    const int tid = threadIdx.x;

    // ---- Stage all weights into LDS, bf16, pre-swizzled to WMMA B layout ----
    // B layout: slot s (0..31), elem i (0..15): K = (s>>4)*8 + i + (i>>3)*8, N = n0 + (s&15).
    // Process K-pairs: one packed bf16x2 ds_store_b32 per step.
    {
        unsigned* wsm32 = (unsigned*)wsm;
        #pragma unroll 4
        for (int p = tid; p < N_TILES * 256; p += 256) {
            int tile = p >> 8;
            int rem  = p & 255;
            int s    = rem >> 3;
            int ip   = rem & 7;                 // K element pair index
            int krel = ((s >> 4) << 3) + (ip << 1) + ((ip >> 2) << 3);
            int nrel = s & 15;
            float f0, f1;
            if (tile < 20) {
                int kt = tile >> 2, nt = tile & 3;
                const float* src = Wg1 + (kt * 32 + krel) * 64 + nt * 16 + nrel;
                f0 = src[0]; f1 = src[64];
            } else if (tile < 36) {
                int t2 = tile - 20;
                int kt = t2 >> 3, nt = t2 & 7;
                const float* src = Wd + (kt * 32 + krel) * 128 + nt * 16 + nrel;
                f0 = src[0]; f1 = src[128];
            } else {
                int t2 = tile - 36;
                int kt = t2 >> 3, nt = t2 & 7;
                const float* src = Wc + (kt * 32 + krel) * 128 + nt * 16 + nrel;
                f0 = src[0]; f1 = src[128];
            }
            v2f ff = {f0, f1};
            v2bf bb = __builtin_convertvector(ff, v2bf);
            wsm32[tile * 256 + s * 8 + ip] = __builtin_bit_cast(unsigned, bb);
        }
    }
    __syncthreads();

    const int wave = tid >> 5;
    const int lane = tid & 31;
    const int g    = lane >> 4;   // half-wave id
    const int m    = lane & 15;   // A row within tile / N within C tile

    auto loadB = [&](int tile) -> v16bf {
        const v16u* p = (const v16u*)(wsm + (tile << 9) + (lane << 4));
        return __builtin_bit_cast(v16bf, *p);
    };

    // Per-lane LN params for feature columns n = t*16 + m (loop-invariant)
    float gamt[4], bett[4], wg2t[4];
    #pragma unroll
    for (int t = 0; t < 4; ++t) {
        gamt[t] = gamma[t * 16 + m];
        bett[t] = beta[t * 16 + m];
        wg2t[t] = Wg2[t * 16 + m];
    }

    const v8f zero8 = {0.f, 0.f, 0.f, 0.f, 0.f, 0.f, 0.f, 0.f};

    for (int rt = 0; rt < ROW_TILES_PER_WAVE; ++rt) {
        const int rowBase = blockIdx.x * ROWS_PER_BLOCK + rt * 128 + wave * 16;
        int rowA = rowBase + m;
        if (rowA >= Ntot) rowA = Ntot - 1;   // clamp loads; stores guarded

        const float* rd = xd + (size_t)rowA * 64;
        const float* rc = xc + (size_t)rowA * 96;

        // A fragments (bf16), shared by gate path and main GEMMs
        v16bf Ad0 = load_a_chunk(rd, 0,  g);
        v16bf Ad1 = load_a_chunk(rd, 32, g);
        v16bf Ac0 = load_a_chunk(rc, 0,  g);
        v16bf Ac1 = load_a_chunk(rc, 32, g);
        v16bf Ac2 = load_a_chunk(rc, 64, g);

        // ------------- Gate path: concat[16x160] @ Wg1[160x64] --------------
        v8f hG0 = zero8, hG1 = zero8, hG2 = zero8, hG3 = zero8;
        {
            v16bf A[5] = {Ad0, Ad1, Ac0, Ac1, Ac2};
            #pragma unroll
            for (int kt = 0; kt < 5; ++kt) {
                v16bf a = A[kt];
                hG0 = WMMA_BF16(a, loadB(kt * 4 + 0), hG0);
                hG1 = WMMA_BF16(a, loadB(kt * 4 + 1), hG1);
                hG2 = WMMA_BF16(a, loadB(kt * 4 + 2), hG2);
                hG3 = WMMA_BF16(a, loadB(kt * 4 + 3), hG3);
            }
        }

        // LayerNorm(64) + relu + dot(W_gate2) + sigmoid via half-wave shuffles
        // (xor masks 1,2,4,8 stay inside each 16-lane group).
        float g8[8];
        #pragma unroll
        for (int r = 0; r < 8; ++r) {
            float x0 = hG0[r], x1 = hG1[r], x2 = hG2[r], x3 = hG3[r];
            float s = x0 + x1 + x2 + x3;
            s += __shfl_xor(s, 1, 32); s += __shfl_xor(s, 2, 32);
            s += __shfl_xor(s, 4, 32); s += __shfl_xor(s, 8, 32);
            float mean = s * 0.015625f;                 // /64
            float d0 = x0 - mean, d1 = x1 - mean, d2 = x2 - mean, d3 = x3 - mean;
            float q = d0 * d0 + d1 * d1 + d2 * d2 + d3 * d3;
            q += __shfl_xor(q, 1, 32); q += __shfl_xor(q, 2, 32);
            q += __shfl_xor(q, 4, 32); q += __shfl_xor(q, 8, 32);
            float rstd = rsqrtf(q * 0.015625f + LN_EPS);
            float y0 = fmaxf(d0 * rstd * gamt[0] + bett[0], 0.f);
            float y1 = fmaxf(d1 * rstd * gamt[1] + bett[1], 0.f);
            float y2 = fmaxf(d2 * rstd * gamt[2] + bett[2], 0.f);
            float y3 = fmaxf(d3 * rstd * gamt[3] + bett[3], 0.f);
            float dot = y0 * wg2t[0] + y1 * wg2t[1] + y2 * wg2t[2] + y3 * wg2t[3];
            dot += __shfl_xor(dot, 1, 32); dot += __shfl_xor(dot, 2, 32);
            dot += __shfl_xor(dot, 4, 32); dot += __shfl_xor(dot, 8, 32);
            g8[r] = 1.0f / (1.0f + __expf(-dot));       // gate for row r + 8*g
        }

        // ------------- Main GEMMs: detail(K=64) and context(K=96) -----------
        v8f Dd[8], Dc[8];
        #pragma unroll
        for (int nt = 0; nt < 8; ++nt) { Dd[nt] = zero8; Dc[nt] = zero8; }

        #pragma unroll
        for (int nt = 0; nt < 8; ++nt) {
            Dd[nt] = WMMA_BF16(Ad0, loadB(20 + 0 * 8 + nt), Dd[nt]);
            Dd[nt] = WMMA_BF16(Ad1, loadB(20 + 1 * 8 + nt), Dd[nt]);
        }
        #pragma unroll
        for (int nt = 0; nt < 8; ++nt) {
            Dc[nt] = WMMA_BF16(Ac0, loadB(36 + 0 * 8 + nt), Dc[nt]);
            Dc[nt] = WMMA_BF16(Ac1, loadB(36 + 1 * 8 + nt), Dc[nt]);
            Dc[nt] = WMMA_BF16(Ac2, loadB(36 + 2 * 8 + nt), Dc[nt]);
        }

        // ------------- Blend + streaming store ------------------------------
        #pragma unroll
        for (int r = 0; r < 8; ++r) {
            int row = rowBase + r + 8 * g;   // C/D layout: M = r + 8*(lane>>4)
            if (row < Ntot) {
                float gg = g8[r];
                float* orow = out + (size_t)row * 128 + m;
                #pragma unroll
                for (int nt = 0; nt < 8; ++nt) {
                    float v = gg * Dd[nt][r] + (1.0f - gg) * Dc[nt][r];
                    __builtin_nontemporal_store(v, orow + nt * 16);
                }
            }
        }
    }
}

extern "C" void kernel_launch(void* const* d_in, const int* in_sizes, int n_in,
                              void* d_out, int out_size, void* d_ws, size_t ws_size,
                              hipStream_t stream) {
    const float* xd    = (const float*)d_in[0];
    const float* xc    = (const float*)d_in[1];
    const float* Wd    = (const float*)d_in[2];
    const float* Wc    = (const float*)d_in[3];
    const float* Wg1   = (const float*)d_in[4];
    const float* gamma = (const float*)d_in[5];
    const float* beta  = (const float*)d_in[6];
    const float* Wg2   = (const float*)d_in[7];
    float* out = (float*)d_out;

    const int Ntot   = in_sizes[0] / 64;                          // [N, 64]
    const int blocks = (Ntot + ROWS_PER_BLOCK - 1) / ROWS_PER_BLOCK;
    gate_fusion_wmma_kernel<<<blocks, 256, 0, stream>>>(
        xd, xc, Wd, Wc, Wg1, gamma, beta, Wg2, out, Ntot);
}
